// DilatedResidualBlock_RanLA_39986145526182
// MI455X (gfx1250) — matrix operand order repaired and checked by reading, not compiled, not run on hardware
//
#include <hip/hip_runtime.h>
#include <math.h>

#define MPTS 16384      // B*N points
#define NPTS 8192       // points per cloud
#define KNB  16         // neighbors per point

typedef __attribute__((ext_vector_type(16))) _Float16 v16h;
typedef __attribute__((ext_vector_type(8)))  float    v8f;
typedef __attribute__((ext_vector_type(4)))  float    v4f;

__device__ __forceinline__ float lrelu_f(float y) { return y >= 0.f ? y : 0.2f * y; }

// ---------------------------------------------------------------------------
// Load one lane's WMMA-A fragment (16 f16 elements) from a float row.
// 16-bit A 16x32 layout: lane L = row m=L%16, half=L/16; element e maps to
// K = (e&7) + 8*half + 16*(e>>3)  ==> two contiguous 8-float runs at
// p = row + kb*32 + 8*half : run0 = p[0..7], run1 = p[16..23].
// All call sites guarantee 16-byte alignment of p.
// ---------------------------------------------------------------------------
__device__ __forceinline__ v16h cvt_a16(const float* __restrict__ p) {
    v4f f0 = *(const v4f*)(p + 0);
    v4f f1 = *(const v4f*)(p + 4);
    v4f f2 = *(const v4f*)(p + 16);
    v4f f3 = *(const v4f*)(p + 20);
    v16h a;
    a[0]  = (_Float16)f0[0]; a[1]  = (_Float16)f0[1];
    a[2]  = (_Float16)f0[2]; a[3]  = (_Float16)f0[3];
    a[4]  = (_Float16)f1[0]; a[5]  = (_Float16)f1[1];
    a[6]  = (_Float16)f1[2]; a[7]  = (_Float16)f1[3];
    a[8]  = (_Float16)f2[0]; a[9]  = (_Float16)f2[1];
    a[10] = (_Float16)f2[2]; a[11] = (_Float16)f2[3];
    a[12] = (_Float16)f3[0]; a[13] = (_Float16)f3[1];
    a[14] = (_Float16)f3[2]; a[15] = (_Float16)f3[3];
    return a;
}

// ---------------------------------------------------------------------------
// Pack a row-major [Kd, Nd] fp32 weight into f16 WMMA-B tiles.
// lane L: n = nt*16 + (L&15), half = L>>4 ; element e: k = kb*32 + 8*half + (e&7) + 16*(e>>3)
// ---------------------------------------------------------------------------
__global__ void k_pack(const float* __restrict__ W, _Float16* __restrict__ dst,
                       int Kd, int Nd, int Kb) {
    int idx = blockIdx.x * blockDim.x + threadIdx.x;
    int total = (Nd >> 4) * Kb * 512;
    if (idx >= total) return;
    int e    = idx & 15;
    int lane = (idx >> 4) & 31;
    int blk  = idx >> 9;
    int kb   = blk % Kb;
    int nt   = blk / Kb;
    int half = lane >> 4;
    int n    = lane & 15;
    int k    = kb * 32 + 8 * half + (e & 7) + 16 * (e >> 3);
    int nn   = nt * 16 + n;
    float v  = (k < Kd) ? W[k * Nd + nn] : 0.f;
    dst[idx] = (_Float16)v;
}

// Fold Linear bias + BatchNorm(inference, eps=1e-6) into y = acc*s + t
__global__ void k_fold(const float* __restrict__ b, const float* __restrict__ g,
                       const float* __restrict__ beta, const float* __restrict__ mean,
                       const float* __restrict__ var, float* __restrict__ s,
                       float* __restrict__ t, int Nd) {
    int n = blockIdx.x * blockDim.x + threadIdx.x;
    if (n >= Nd) return;
    float sv = g[n] * rsqrtf(var[n] + 1e-6f);
    s[n] = sv;
    t[n] = (b[n] - mean[n]) * sv + beta[n];
}

// ---------------------------------------------------------------------------
// Brute-force kNN (k=16, incl. self) per cloud of 8192. One thread per point,
// LDS tile of 256 candidate positions, unrolled 16-entry insertion list.
// ---------------------------------------------------------------------------
__global__ void k_knn(const float* __restrict__ pos, int* __restrict__ nbr) {
    __shared__ float tp[256 * 3];
    int i    = blockIdx.x * 256 + threadIdx.x;
    int base = (i >> 13) << 13;              // batch start (8192 per cloud)
    float px = pos[i * 3 + 0], py = pos[i * 3 + 1], pz = pos[i * 3 + 2];
    float bd[16];
    int   bi[16];
#pragma unroll
    for (int t = 0; t < 16; ++t) { bd[t] = 3.4e38f; bi[t] = base; }
    for (int tile = 0; tile < NPTS; tile += 256) {
        __syncthreads();
        int j0 = base + tile + threadIdx.x;
        tp[threadIdx.x * 3 + 0] = pos[j0 * 3 + 0];
        tp[threadIdx.x * 3 + 1] = pos[j0 * 3 + 1];
        tp[threadIdx.x * 3 + 2] = pos[j0 * 3 + 2];
        __syncthreads();
        for (int jj = 0; jj < 256; ++jj) {
            float dx = tp[jj * 3 + 0] - px;
            float dy = tp[jj * 3 + 1] - py;
            float dz = tp[jj * 3 + 2] - pz;
            float d2 = dx * dx + dy * dy + dz * dz;
            if (d2 < bd[15]) {
                bd[15] = d2; bi[15] = base + tile + jj;
#pragma unroll
                for (int t = 15; t > 0; --t) {
                    if (bd[t] < bd[t - 1]) {
                        float td = bd[t]; bd[t] = bd[t - 1]; bd[t - 1] = td;
                        int   ti = bi[t]; bi[t] = bi[t - 1]; bi[t - 1] = ti;
                    }
                }
            }
        }
    }
#pragma unroll
    for (int t = 0; t < 16; ++t) nbr[i * KNB + t] = bi[t];
}

// ---------------------------------------------------------------------------
// Generic WMMA GEMM: out[M,Nd] = epilogue(A[M,Kd] @ Wp); one wave per 16x16
// tile. Kd MUST be a multiple of 32 (guaranteed by buffer strides).
// ---------------------------------------------------------------------------
__global__ void k_gemm(const float* __restrict__ A, const _Float16* __restrict__ Wp,
                       const float* __restrict__ s, const float* __restrict__ t,
                       const float* __restrict__ add, float* __restrict__ out,
                       int Mrows, int Kd, int Kb, int Nd, int act) {
    const int lane = threadIdx.x & 31;
    const int wave = threadIdx.x >> 5;
    const int half = lane >> 4;
    const int ml   = lane & 15;
    const int mtiles = Mrows >> 4;
    int tile = blockIdx.x * 4 + wave;
    int tm = tile % mtiles;
    int tn = tile / mtiles;
    const v16h* Bp = (const v16h*)Wp;
    const float* Arow = A + (size_t)(tm * 16 + ml) * Kd + 8 * half;
    v8f acc = {};
    for (int kb = 0; kb < Kb; ++kb) {
        v16h a = cvt_a16(Arow + kb * 32);
        v16h b = Bp[(tn * Kb + kb) * 32 + lane];
        acc = __builtin_amdgcn_wmma_f32_16x16x32_f16(false, a, false, b, (short)0,
                                                     acc, false, false);
    }
    const int ng = tn * 16 + ml;
    const float sv = s[ng];
    const float tv = t[ng];
#pragma unroll
    for (int r = 0; r < 8; ++r) {
        int mg = tm * 16 + r + 8 * half;
        float y = acc[r] * sv + tv;
        if (add) y += add[mg * Nd + ng];
        if (act) y = lrelu_f(y);
        out[mg * Nd + ng] = y;
    }
}

// ---------------------------------------------------------------------------
// LFA: one wave per point. Builds rel/edge/local tiles (M=16 neighbors) in
// LDS, runs enc/dgcnn/att via WMMA, softmax over neighbors via shfl_xor(16),
// writes attention- and raw-aggregated features to workspace.
// C2 = input channel count (c/2): 16 for lfa1, 32 for lfa2.
// ---------------------------------------------------------------------------
template <int C2>
__global__ void k_lfa(const float* __restrict__ x, const float* __restrict__ pos,
                      const int* __restrict__ nbr,
                      const _Float16* __restrict__ Wenc, const float* __restrict__ senc,
                      const float* __restrict__ tenc,
                      const _Float16* __restrict__ Wdg, const float* __restrict__ sdg,
                      const float* __restrict__ tdg,
                      const _Float16* __restrict__ Watt,
                      float* __restrict__ agg_att, float* __restrict__ agg_raw) {
    constexpr int C   = 2 * C2;             // 32 / 64
    constexpr int LW  = 2 * C;              // local width: 64 / 128
    constexpr int DKB = (3 * C2 + 31) / 32; // edge k-blocks: 2 / 3
    constexpr int AKB = LW / 32;            // att k-blocks: 2 / 4
    constexpr int ANT = LW / 16;            // att n-tiles: 4 / 8
    constexpr int DNT = C / 16;             // dgcnn n-tiles: 2 / 4
    constexpr int ENT = C2 / 16;            // enc n-tiles: 1 / 2
    constexpr int AS = 132, DS = 66, XS = 34;

    __shared__ float Abuf[2][16 * AS];
    __shared__ float dgb[2][16 * DS];
    __shared__ float lseb[2][16 * XS];
    __shared__ float xjb[2][16 * XS];
    __shared__ int   nbl[2][16];
    __shared__ float pjb[2][48];

    const int lane = threadIdx.x & 31;
    const int w    = threadIdx.x >> 5;
    const int half = lane >> 4;
    const int ml   = lane & 15;
    const int i    = blockIdx.x * 2 + w;

    if (lane < 16) {
        int j = nbr[i * KNB + lane];
        nbl[w][lane] = j;
        pjb[w][lane * 3 + 0] = pos[j * 3 + 0];
        pjb[w][lane * 3 + 1] = pos[j * 3 + 1];
        pjb[w][lane * 3 + 2] = pos[j * 3 + 2];
    }
    __builtin_amdgcn_wave_barrier();

    const float pix = pos[i * 3 + 0], piy = pos[i * 3 + 1], piz = pos[i * 3 + 2];
    const float* Arow = &Abuf[w][ml * AS + 8 * half];   // lane's A-fragment base

    // gather neighbor features
    for (int idx = lane; idx < 16 * C2; idx += 32) {
        int row = idx / C2, col = idx % C2;
        xjb[w][row * XS + col] = x[nbl[w][row] * C2 + col];
    }

    // rel = [pos_i | pos_j | diff | dist | 0-pad] -> [16 x 32]
    for (int idx = lane; idx < 16 * 32; idx += 32) {
        int row = idx >> 5, col = idx & 31;
        float dx = pjb[w][row * 3 + 0] - pix;
        float dy = pjb[w][row * 3 + 1] - piy;
        float dz = pjb[w][row * 3 + 2] - piz;
        float v = 0.f;
        if (col < 3)       v = (col == 0 ? pix : col == 1 ? piy : piz);
        else if (col < 6)  v = pjb[w][row * 3 + col - 3];
        else if (col < 9)  v = (col == 6 ? dx : col == 7 ? dy : dz);
        else if (col == 9) v = sqrtf(fmaxf(dx * dx + dy * dy + dz * dz, 1e-12f));
        Abuf[w][row * AS + col] = v;
    }
    __builtin_amdgcn_wave_barrier();

    // enc: [16x10(pad32)] @ [10 x C2] -> lse
    {
        v16h a = cvt_a16(Arow);
        const v16h* Bp = (const v16h*)Wenc;
#pragma unroll
        for (int nt = 0; nt < ENT; ++nt) {
            v8f acc = {};
            v16h b = Bp[nt * 32 + lane];
            acc = __builtin_amdgcn_wmma_f32_16x16x32_f16(false, a, false, b, (short)0,
                                                         acc, false, false);
#pragma unroll
            for (int r = 0; r < 8; ++r) {
                int mr = r + 8 * half, n = nt * 16 + ml;
                lseb[w][mr * XS + n] = lrelu_f(acc[r] * senc[n] + tenc[n]);
            }
        }
    }
    __builtin_amdgcn_wave_barrier();

    // edge = [x_i | x_j | x_j - x_i | 0-pad] -> [16 x DKB*32]
    for (int idx = lane; idx < 16 * (DKB * 32); idx += 32) {
        int row = idx / (DKB * 32), col = idx % (DKB * 32);
        float v = 0.f;
        if (col < C2)           v = x[i * C2 + col];
        else if (col < 2 * C2)  v = xjb[w][row * XS + col - C2];
        else if (col < 3 * C2)  v = xjb[w][row * XS + col - 2 * C2] - x[i * C2 + col - 2 * C2];
        Abuf[w][row * AS + col] = v;
    }
    __builtin_amdgcn_wave_barrier();

    // dgcnn: [16 x 3*C2] @ [3*C2 x C] -> dg
    {
        const v16h* Bp = (const v16h*)Wdg;
#pragma unroll
        for (int nt = 0; nt < DNT; ++nt) {
            v8f acc = {};
#pragma unroll
            for (int kb = 0; kb < DKB; ++kb) {
                v16h a = cvt_a16(Arow + kb * 32);
                v16h b = Bp[(nt * DKB + kb) * 32 + lane];
                acc = __builtin_amdgcn_wmma_f32_16x16x32_f16(false, a, false, b, (short)0,
                                                             acc, false, false);
            }
#pragma unroll
            for (int r = 0; r < 8; ++r) {
                int mr = r + 8 * half, n = nt * 16 + ml;
                dgb[w][mr * DS + n] = lrelu_f(acc[r] * sdg[n] + tdg[n]);
            }
        }
    }
    __builtin_amdgcn_wave_barrier();

    // local = [dg | x_j | lse] -> [16 x LW]
    for (int idx = lane; idx < 16 * LW; idx += 32) {
        int row = idx / LW, col = idx % LW;
        float v;
        if (col < C)            v = dgb[w][row * DS + col];
        else if (col < C + C2)  v = xjb[w][row * XS + col - C];
        else                    v = lseb[w][row * XS + col - C - C2];
        Abuf[w][row * AS + col] = v;
    }
    __builtin_amdgcn_wave_barrier();

    // att logits + column softmax over 16 neighbors + aggregation
    {
        const v16h* Bp = (const v16h*)Watt;
#pragma unroll
        for (int nt = 0; nt < ANT; ++nt) {
            v8f acc = {};
#pragma unroll
            for (int kb = 0; kb < AKB; ++kb) {
                v16h a = cvt_a16(Arow + kb * 32);
                v16h b = Bp[(nt * AKB + kb) * 32 + lane];
                acc = __builtin_amdgcn_wmma_f32_16x16x32_f16(false, a, false, b, (short)0,
                                                             acc, false, false);
            }
            // column n lives in lanes L and L^16 (8 rows each)
            float mx = acc[0];
#pragma unroll
            for (int r = 1; r < 8; ++r) mx = fmaxf(mx, acc[r]);
            mx = fmaxf(mx, __shfl_xor(mx, 16, 32));
            float ex[8], sum = 0.f;
#pragma unroll
            for (int r = 0; r < 8; ++r) { ex[r] = __expf(acc[r] - mx); sum += ex[r]; }
            sum += __shfl_xor(sum, 16, 32);
            float inv = 1.f / sum;
            int n = nt * 16 + ml;
            float pa = 0.f, pr = 0.f;
#pragma unroll
            for (int r = 0; r < 8; ++r) {
                float lv = Abuf[w][(r + 8 * half) * AS + n];
                pa += ex[r] * inv * lv;
                pr += lv;
            }
            pa += __shfl_xor(pa, 16, 32);
            pr += __shfl_xor(pr, 16, 32);
            if (half == 0) {
                agg_att[i * LW + n] = pa;
                if (agg_raw) agg_raw[i * LW + n] = pr;
            }
        }
    }
}

// ---------------------------------------------------------------------------
extern "C" void kernel_launch(void* const* d_in, const int* in_sizes, int n_in,
                              void* d_out, int out_size, void* d_ws, size_t ws_size,
                              hipStream_t stream) {
    (void)in_sizes; (void)n_in; (void)out_size; (void)ws_size;
    // jax tree-leaf (sorted-key) flattening of setup_inputs():
    const float* Watt1_f = (const float*)d_in[0];
    const float* dg1_beta = (const float*)d_in[1], *dg1_g = (const float*)d_in[2],
               * dg1_mean = (const float*)d_in[3], *dg1_var = (const float*)d_in[4],
               * dg1_b = (const float*)d_in[5], *dg1_w = (const float*)d_in[6];
    const float* en1_beta = (const float*)d_in[7], *en1_g = (const float*)d_in[8],
               * en1_mean = (const float*)d_in[9], *en1_var = (const float*)d_in[10],
               * en1_b = (const float*)d_in[11], *en1_w = (const float*)d_in[12];
    const float* po1_beta = (const float*)d_in[13], *po1_g = (const float*)d_in[14],
               * po1_mean = (const float*)d_in[15], *po1_var = (const float*)d_in[16],
               * po1_b = (const float*)d_in[17], *po1_w = (const float*)d_in[18];
    const float* ra1_beta = (const float*)d_in[19], *ra1_g = (const float*)d_in[20],
               * ra1_mean = (const float*)d_in[21], *ra1_var = (const float*)d_in[22],
               * ra1_b = (const float*)d_in[23], *ra1_w = (const float*)d_in[24];
    const float* Watt2_f = (const float*)d_in[25];
    const float* dg2_beta = (const float*)d_in[26], *dg2_g = (const float*)d_in[27],
               * dg2_mean = (const float*)d_in[28], *dg2_var = (const float*)d_in[29],
               * dg2_b = (const float*)d_in[30], *dg2_w = (const float*)d_in[31];
    const float* en2_beta = (const float*)d_in[32], *en2_g = (const float*)d_in[33],
               * en2_mean = (const float*)d_in[34], *en2_var = (const float*)d_in[35],
               * en2_b = (const float*)d_in[36], *en2_w = (const float*)d_in[37];
    const float* po2_beta = (const float*)d_in[38], *po2_g = (const float*)d_in[39],
               * po2_mean = (const float*)d_in[40], *po2_var = (const float*)d_in[41],
               * po2_b = (const float*)d_in[42], *po2_w = (const float*)d_in[43];
    // d_in[44..49]: lfa2.raw (unused: return_raw=False for lfa2)
    const float* m1_beta = (const float*)d_in[50], *m1_g = (const float*)d_in[51],
               * m1_mean = (const float*)d_in[52], *m1_var = (const float*)d_in[53],
               * m1_b = (const float*)d_in[54], *m1_w = (const float*)d_in[55];
    const float* m2_beta = (const float*)d_in[56], *m2_g = (const float*)d_in[57],
               * m2_mean = (const float*)d_in[58], *m2_var = (const float*)d_in[59],
               * m2_b = (const float*)d_in[60], *m2_w = (const float*)d_in[61];
    const float* sc_beta = (const float*)d_in[62], *sc_g = (const float*)d_in[63],
               * sc_mean = (const float*)d_in[64], *sc_var = (const float*)d_in[65],
               * sc_b = (const float*)d_in[66], *sc_w = (const float*)d_in[67];
    const float* pos = (const float*)d_in[68];
    const float* x   = (const float*)d_in[69];

    float* out   = (float*)d_out;
    float* out_fp  = out;                                // [M,128]
    float* out_pos = out + MPTS * 128;                   // [M,3]
    float* out_rec = out + MPTS * 128 + MPTS * 3;        // [M,128]

    // workspace carve-up (deterministic)
    char* ws = (char*)d_ws;
    size_t cur = 0;
    auto alloc = [&](size_t bytes) -> void* {
        void* p = ws + cur;
        cur = (cur + bytes + 255) & ~(size_t)255;
        return p;
    };
    int*   nbr     = (int*)  alloc(MPTS * KNB * 4);
    float* h0      = (float*)alloc(MPTS * 16 * 4);
    float* shortb  = (float*)alloc(MPTS * 128 * 4);
    float* h1      = (float*)alloc(MPTS * 32 * 4);
    float* h2      = (float*)alloc(MPTS * 64 * 4);
    float* aggA    = (float*)alloc(MPTS * 128 * 4);
    float* aggR    = (float*)alloc(MPTS * 128 * 4);

    auto pack = [&](const float* Wf, int Kd, int Nd) -> _Float16* {
        int Kb = (Kd + 31) / 32;
        int total = (Nd >> 4) * Kb * 512;
        _Float16* dst = (_Float16*)alloc((size_t)total * sizeof(_Float16));
        k_pack<<<(total + 255) / 256, 256, 0, stream>>>(Wf, dst, Kd, Nd, Kb);
        return dst;
    };
    auto fold = [&](const float* b, const float* g, const float* be, const float* mn,
                    const float* vr, int Nd, float** sp, float** tp) {
        *sp = (float*)alloc(Nd * 4);
        *tp = (float*)alloc(Nd * 4);
        k_fold<<<(Nd + 63) / 64, 64, 0, stream>>>(b, g, be, mn, vr, *sp, *tp, Nd);
    };

    // pack weights + fold BN
    _Float16* Wm1  = pack(m1_w, 32, 16);
    _Float16* Wsc  = pack(sc_w, 32, 128);
    _Float16* Wm2  = pack(m2_w, 64, 128);
    _Float16* Wen1 = pack(en1_w, 10, 16);
    _Float16* Wdg1 = pack(dg1_w, 48, 32);
    _Float16* Wat1 = pack(Watt1_f, 64, 64);
    _Float16* Wpo1 = pack(po1_w, 64, 32);
    _Float16* Wra1 = pack(ra1_w, 64, 128);
    _Float16* Wen2 = pack(en2_w, 10, 32);
    _Float16* Wdg2 = pack(dg2_w, 96, 64);
    _Float16* Wat2 = pack(Watt2_f, 128, 128);
    _Float16* Wpo2 = pack(po2_w, 128, 64);

    float *s_m1, *t_m1, *s_sc, *t_sc, *s_m2, *t_m2;
    float *s_en1, *t_en1, *s_dg1, *t_dg1, *s_po1, *t_po1, *s_ra1, *t_ra1;
    float *s_en2, *t_en2, *s_dg2, *t_dg2, *s_po2, *t_po2;
    fold(m1_b, m1_g, m1_beta, m1_mean, m1_var, 16, &s_m1, &t_m1);
    fold(sc_b, sc_g, sc_beta, sc_mean, sc_var, 128, &s_sc, &t_sc);
    fold(m2_b, m2_g, m2_beta, m2_mean, m2_var, 128, &s_m2, &t_m2);
    fold(en1_b, en1_g, en1_beta, en1_mean, en1_var, 16, &s_en1, &t_en1);
    fold(dg1_b, dg1_g, dg1_beta, dg1_mean, dg1_var, 32, &s_dg1, &t_dg1);
    fold(po1_b, po1_g, po1_beta, po1_mean, po1_var, 32, &s_po1, &t_po1);
    fold(ra1_b, ra1_g, ra1_beta, ra1_mean, ra1_var, 128, &s_ra1, &t_ra1);
    fold(en2_b, en2_g, en2_beta, en2_mean, en2_var, 32, &s_en2, &t_en2);
    fold(dg2_b, dg2_g, dg2_beta, dg2_mean, dg2_var, 64, &s_dg2, &t_dg2);
    fold(po2_b, po2_g, po2_beta, po2_mean, po2_var, 64, &s_po2, &t_po2);

    // kNN
    k_knn<<<MPTS / 256, 256, 0, stream>>>(pos, nbr);

    // mlp1: x[M,32] -> h0[M,16] (lrelu) ; shortcut: x -> short[M,128] (no act)
    k_gemm<<<(MPTS / 16) * (16 / 16) / 4, 128, 0, stream>>>(x, Wm1, s_m1, t_m1, nullptr,
                                                            h0, MPTS, 32, 1, 16, 1);
    k_gemm<<<(MPTS / 16) * (128 / 16) / 4, 128, 0, stream>>>(x, Wsc, s_sc, t_sc, nullptr,
                                                             shortb, MPTS, 32, 1, 128, 0);
    // LFA1 (c=32): h0 -> agg[M,64]
    k_lfa<16><<<MPTS / 2, 64, 0, stream>>>(h0, pos, nbr, Wen1, s_en1, t_en1, Wdg1, s_dg1,
                                           t_dg1, Wat1, aggA, aggR);
    // post1: aggA[M,64] -> h1[M,32] ; raw1: aggR[M,64] -> rec[M,128]
    k_gemm<<<(MPTS / 16) * (32 / 16) / 4, 128, 0, stream>>>(aggA, Wpo1, s_po1, t_po1,
                                                            nullptr, h1, MPTS, 64, 2, 32, 1);
    k_gemm<<<(MPTS / 16) * (128 / 16) / 4, 128, 0, stream>>>(aggR, Wra1, s_ra1, t_ra1,
                                                             nullptr, out_rec, MPTS, 64, 2,
                                                             128, 1);
    // LFA2 (c=64): h1 -> aggA[M,128]
    k_lfa<32><<<MPTS / 2, 64, 0, stream>>>(h1, pos, nbr, Wen2, s_en2, t_en2, Wdg2, s_dg2,
                                           t_dg2, Wat2, aggA, nullptr);
    // post2: aggA[M,128] -> h2[M,64]
    k_gemm<<<(MPTS / 16) * (64 / 16) / 4, 128, 0, stream>>>(aggA, Wpo2, s_po2, t_po2,
                                                            nullptr, h2, MPTS, 128, 4, 64, 1);
    // mlp2 + residual + lrelu: f_p = lrelu(bn(h2@W+b) + short)
    k_gemm<<<(MPTS / 16) * (128 / 16) / 4, 128, 0, stream>>>(h2, Wm2, s_m2, t_m2, shortb,
                                                             out_fp, MPTS, 64, 2, 128, 1);
    // pos passthrough
    hipMemcpyAsync(out_pos, pos, MPTS * 3 * sizeof(float), hipMemcpyDeviceToDevice, stream);
}
